// GATMODEL_27212912787480
// MI455X (gfx1250) — matrix operand order repaired and checked by reading, not compiled, not run on hardware
//
#include <hip/hip_runtime.h>
#include <hip/hip_bf16.h>
#include <math.h>

typedef __attribute__((ext_vector_type(16))) _Float16 v16h;
typedef __attribute__((ext_vector_type(8)))  float    v8f;

#define SLOPE 0.2f

// ---------------------------------------------------------------------------
// Order-preserving float<->uint encoding for atomic max on floats.
// enc is strictly monotonic; memory initialized to 0 sits below enc(-inf).
// ---------------------------------------------------------------------------
__device__ __forceinline__ unsigned enc_f32(float f) {
    unsigned u = __float_as_uint(f);
    return (u >> 31) ? ~u : (u | 0x80000000u);
}
__device__ __forceinline__ float dec_f32(unsigned u) {
    return (u >> 31) ? __uint_as_float(u & 0x7FFFFFFFu) : __uint_as_float(~u);
}
__device__ __forceinline__ float lrelu(float v) { return v > 0.f ? v : SLOPE * v; }

// ---------------------------------------------------------------------------
// Pack X (f32 [n_rows,128]) into f16 A-fragments in WMMA register layout.
// Fragment index = (row_tile*4 + ktile)*32 + lane; each fragment = 16 f16 (32B).
// A layout (16-bit A 16x32): lane=(half,m); VGPR i<4 -> K=2i+8*half,
// i>=4 -> K=16+2(i-4)+8*half; element pair (2i,2i+1) = K, K+1.
// ---------------------------------------------------------------------------
__global__ __launch_bounds__(256) void gat5_pack_A(
    const float* __restrict__ X, _Float16* __restrict__ Afrag, int n_rows)
{
    const int tid = blockIdx.x * blockDim.x + threadIdx.x;
    const int n_rt = (n_rows + 15) >> 4;
    if (tid >= n_rt * 4 * 32) return;
    const int lane = tid & 31;
    const int t    = (tid >> 5) & 3;     // ktile
    const int rt   = tid >> 7;           // row tile
    const int half = lane >> 4, l16 = lane & 15;
    int m = rt * 16 + l16;
    if (m >= n_rows) m = n_rows - 1;     // clamped; GEMM stores are guarded
    const float* __restrict__ xrow = X + (size_t)m * 128 + t * 32;
    v16h a;
    #pragma unroll
    for (int i = 0; i < 8; ++i) {
        const int kl = 2 * i + (i >= 4 ? 8 : 0) + 8 * half;
        const float2 v = *(const float2*)(xrow + kl);
        a[2 * i]     = (_Float16)v.x;
        a[2 * i + 1] = (_Float16)v.y;
    }
    *(v16h*)(Afrag + (size_t)tid * 16) = a;
}

// ---------------------------------------------------------------------------
// Pack W (f32 [128,128]) into f16 B-fragments in WMMA register layout.
// Fragment index = (col_tile*4 + ktile)*32 + lane.
// B layout (16-bit B 32x16): lane=(half,n); VGPR v<4 -> K=2v+8*half,
// v>=4 -> K=16+2(v-4)+8*half; holds rows K, K+1 at column n.
// ---------------------------------------------------------------------------
__global__ __launch_bounds__(256) void gat5_pack_B(
    const float* __restrict__ W, _Float16* __restrict__ Bfrag)
{
    const int tid = blockIdx.x * blockDim.x + threadIdx.x;
    if (tid >= 8 * 4 * 32) return;
    const int lane = tid & 31;
    const int t    = (tid >> 5) & 3;     // ktile
    const int ct   = tid >> 7;           // col tile
    const int half = lane >> 4, l16 = lane & 15;
    const int n = ct * 16 + l16;
    v16h b;
    #pragma unroll
    for (int v = 0; v < 8; ++v) {
        const int kb = t * 32 + 2 * v + (v >= 4 ? 8 : 0) + 8 * half;
        b[2 * v]     = (_Float16)W[(size_t)kb * 128 + n];
        b[2 * v + 1] = (_Float16)W[(size_t)(kb + 1) * 128 + n];
    }
    *(v16h*)(Bfrag + (size_t)tid * 16) = b;
}

// ---------------------------------------------------------------------------
// GEMM h = X @ W via v_wmma_f32_16x16x32_f16 on pre-packed fragments, with
// fused attention-coefficient epilogue.
// One wave -> one 16x16 tile: 4 K-steps, each 4x global_load_b128 + 1 WMMA.
// Column tile == head (C == 16): epilogue forms asrc/adst[row,head] via
// 16-lane butterfly reduction of acc[r] * a_vec (no extra pass over H).
// ---------------------------------------------------------------------------
__global__ __launch_bounds__(256) void gat5_gemm_wmma(
    const _Float16* __restrict__ Afrag, const _Float16* __restrict__ Bfrag,
    const float* __restrict__ a_src, const float* __restrict__ a_dst,
    float* __restrict__ H_out, float* __restrict__ asrc_out,
    float* __restrict__ adst_out, int n_rows)
{
    const int lane = threadIdx.x & 31;
    const int wave = threadIdx.x >> 5;   // column tile == head
    const int rt   = blockIdx.x;         // row tile

    const v16h* __restrict__ ap = (const v16h*)Afrag + (size_t)(rt * 4) * 32 + lane;
    const v16h* __restrict__ bp = (const v16h*)Bfrag + (size_t)(wave * 4) * 32 + lane;

    v8f acc = {};
    #pragma unroll
    for (int t = 0; t < 4; ++t) {
        const v16h a = ap[t * 32];
        const v16h b = bp[t * 32];
        acc = __builtin_amdgcn_wmma_f32_16x16x32_f16(
            /*neg_a=*/false, a, /*neg_b=*/false, b,
            /*c_mod=*/(short)0, acc, /*reuse_a=*/false, /*reuse_b=*/false);
    }

    // C/D layout: VGPR r -> row (rt*16 + r + 8*half), col = wave*16 + (lane&15)
    const int half = lane >> 4, l16 = lane & 15;
    const bool full_tile = (rt * 16 + 16 <= n_rows);

    float* __restrict__ hbase =
        H_out + (size_t)(rt * 16 + 8 * half) * 128 + wave * 16 + l16;
    if (full_tile) {
        #pragma unroll
        for (int r = 0; r < 8; ++r) hbase[(size_t)r * 128] = acc[r];
    } else {
        #pragma unroll
        for (int r = 0; r < 8; ++r)
            if (rt * 16 + r + 8 * half < n_rows) hbase[(size_t)r * 128] = acc[r];
    }

    // Fused attention coefficients: asrc[row,head] = <h[row,head,:], a_src[head,:]>
    const float asv = a_src[wave * 16 + l16];
    const float adv = a_dst[wave * 16 + l16];
    float sr[8], dr[8];
    #pragma unroll
    for (int r = 0; r < 8; ++r) {
        float s = acc[r] * asv;
        float d = acc[r] * adv;
        #pragma unroll
        for (int mk = 8; mk >= 1; mk >>= 1) {   // stays within 16-lane half-group
            s += __shfl_xor(s, mk);
            d += __shfl_xor(d, mk);
        }
        sr[r] = s; dr[r] = d;
    }
    if (l16 == 0) {   // one lane per half-group writes the 8 rows of its half
        #pragma unroll
        for (int r = 0; r < 8; ++r) {
            const int row = rt * 16 + r + 8 * half;
            if (full_tile || row < n_rows) {
                asrc_out[(size_t)row * 8 + wave] = sr[r];
                adst_out[(size_t)row * 8 + wave] = dr[r];
            }
        }
    }
}

// ---------------------------------------------------------------------------
// Init: accumulator seeded with bias, emax/denom cleared.
// ---------------------------------------------------------------------------
__global__ __launch_bounds__(256) void gat5_init(
    float* __restrict__ out_agg, const float* __restrict__ bias,
    unsigned* __restrict__ emax, float* __restrict__ denom, int n_nodes)
{
    const int tid = blockIdx.x * blockDim.x + threadIdx.x;
    if (tid < n_nodes * 128) out_agg[tid] = bias[tid & 127];
    if (tid < n_nodes * 8) { emax[tid] = 0u; denom[tid] = 0.f; }
}

// ---------------------------------------------------------------------------
// Edge pass 1 (one thread per edge, all 8 heads):
// e = leaky_relu(asrc[src]+adst[dst]); segment max into emax[dst].
// ---------------------------------------------------------------------------
__global__ __launch_bounds__(256) void gat5_edge_max(
    const int* __restrict__ src, const int* __restrict__ dst,
    const float* __restrict__ asrc, const float* __restrict__ adst,
    float* __restrict__ ebuf, unsigned* __restrict__ emax, int n_edges)
{
    const int e = blockIdx.x * blockDim.x + threadIdx.x;
    if (e >= n_edges) return;
    const int s = src[e], d = dst[e];
    const float4* __restrict__ as = (const float4*)(asrc + (size_t)s * 8);
    const float4* __restrict__ ad = (const float4*)(adst + (size_t)d * 8);
    float4* __restrict__ eb = (float4*)(ebuf + (size_t)e * 8);
    unsigned* __restrict__ em = emax + (size_t)d * 8;
    #pragma unroll
    for (int q = 0; q < 2; ++q) {
        const float4 a = as[q], b = ad[q];
        float4 v;
        v.x = lrelu(a.x + b.x); v.y = lrelu(a.y + b.y);
        v.z = lrelu(a.z + b.z); v.w = lrelu(a.w + b.w);
        eb[q] = v;
        atomicMax(em + 4 * q + 0, enc_f32(v.x));
        atomicMax(em + 4 * q + 1, enc_f32(v.y));
        atomicMax(em + 4 * q + 2, enc_f32(v.z));
        atomicMax(em + 4 * q + 3, enc_f32(v.w));
    }
}

// ---------------------------------------------------------------------------
// Edge pass 2 (one thread per edge): ex = exp(e - emax[dst]); sum into denom.
// ---------------------------------------------------------------------------
__global__ __launch_bounds__(256) void gat5_edge_exp(
    const int* __restrict__ dst, const unsigned* __restrict__ emax,
    float* __restrict__ ebuf, float* __restrict__ denom, int n_edges)
{
    const int e = blockIdx.x * blockDim.x + threadIdx.x;
    if (e >= n_edges) return;
    const int d = dst[e];
    const uint4* __restrict__ em = (const uint4*)(emax + (size_t)d * 8);
    float4* __restrict__ eb = (float4*)(ebuf + (size_t)e * 8);
    float* __restrict__ dn = denom + (size_t)d * 8;
    #pragma unroll
    for (int q = 0; q < 2; ++q) {
        const uint4 mu = em[q];
        float4 v = eb[q];
        v.x = __expf(v.x - dec_f32(mu.x));
        v.y = __expf(v.y - dec_f32(mu.y));
        v.z = __expf(v.z - dec_f32(mu.z));
        v.w = __expf(v.w - dec_f32(mu.w));
        eb[q] = v;
        atomicAdd(dn + 4 * q + 0, v.x);
        atomicAdd(dn + 4 * q + 1, v.y);
        atomicAdd(dn + 4 * q + 2, v.z);
        atomicAdd(dn + 4 * q + 3, v.w);
    }
}

// ---------------------------------------------------------------------------
// Edge pass 3 (one thread per (edge,head)): alpha = ex / (denom[dst]+1e-16);
// emit alpha; scatter out[dst,h,:] += alpha * h[src,h,:]  (16 f32 atomics).
// ---------------------------------------------------------------------------
__global__ __launch_bounds__(256) void gat5_edge_scatter(
    const int* __restrict__ src, const int* __restrict__ dst,
    const float* __restrict__ ebuf, const float* __restrict__ denom,
    const float* __restrict__ H, float* __restrict__ out_agg,
    float* __restrict__ alpha_out, int n_edges)
{
    const int tid = blockIdx.x * blockDim.x + threadIdx.x;
    if (tid >= n_edges * 8) return;
    const int e = tid >> 3, h = tid & 7;
    const int s = src[e], d = dst[e];
    const float alpha = ebuf[tid] / (denom[(size_t)d * 8 + h] + 1e-16f);
    alpha_out[tid] = alpha;
    const float4* __restrict__ hp = (const float4*)(H + (size_t)s * 128 + h * 16);
    float* __restrict__ op = out_agg + (size_t)d * 128 + h * 16;
    #pragma unroll
    for (int q = 0; q < 4; ++q) {
        const float4 v = hp[q];
        atomicAdd(op + 4 * q + 0, v.x * alpha);
        atomicAdd(op + 4 * q + 1, v.y * alpha);
        atomicAdd(op + 4 * q + 2, v.z * alpha);
        atomicAdd(op + 4 * q + 3, v.w * alpha);
    }
}

// ---------------------------------------------------------------------------
// Host-side: one GAT layer = 7 kernel launches on `stream`.
// ---------------------------------------------------------------------------
static void run_gat_layer(const float* X, const float* W, const float* a_src,
                          const float* a_dst, const float* bias,
                          const int* src, const int* dst,
                          _Float16* Afrag, _Float16* Bfrag,
                          float* hbuf, float* asrc, float* adst,
                          unsigned* emax, float* denom, float* ebuf,
                          float* out_agg, float* alpha_out,
                          int N, int E, hipStream_t stream)
{
    const int TB = 256;
    const int row_tiles = (N + 15) / 16;
    gat5_pack_A<<<(row_tiles * 128 + TB - 1) / TB, TB, 0, stream>>>(X, Afrag, N);
    gat5_pack_B<<<4, TB, 0, stream>>>(W, Bfrag);
    gat5_gemm_wmma<<<row_tiles, TB, 0, stream>>>(Afrag, Bfrag, a_src, a_dst,
                                                 hbuf, asrc, adst, N);
    gat5_init<<<(N * 128 + TB - 1) / TB, TB, 0, stream>>>(out_agg, bias, emax, denom, N);
    gat5_edge_max<<<(E + TB - 1) / TB, TB, 0, stream>>>(src, dst, asrc, adst, ebuf, emax, E);
    gat5_edge_exp<<<(E + TB - 1) / TB, TB, 0, stream>>>(dst, emax, ebuf, denom, E);
    gat5_edge_scatter<<<(E * 8 + TB - 1) / TB, TB, 0, stream>>>(src, dst, ebuf, denom, hbuf,
                                                               out_agg, alpha_out, E);
}

extern "C" void kernel_launch(void* const* d_in, const int* in_sizes, int n_in,
                              void* d_out, int out_size, void* d_ws, size_t ws_size,
                              hipStream_t stream)
{
    // setup_inputs order: x, W1, a_src1, a_dst1, b1, W2, a_src2, a_dst2, b2, edge_index
    const float* x      = (const float*)d_in[0];
    const float* W1     = (const float*)d_in[1];
    const float* a_src1 = (const float*)d_in[2];
    const float* a_dst1 = (const float*)d_in[3];
    const float* b1     = (const float*)d_in[4];
    const float* W2     = (const float*)d_in[5];
    const float* a_src2 = (const float*)d_in[6];
    const float* a_dst2 = (const float*)d_in[7];
    const float* b2     = (const float*)d_in[8];
    const int*   ei     = (const int*)d_in[9];

    const int N = in_sizes[0] / 128;    // 50000
    const int E = in_sizes[9] / 2;      // 800000
    const int* src = ei;
    const int* dst = ei + E;

    // Workspace carve-up (~96 MB total)
    float*    hbuf  = (float*)d_ws;                       // [N,128] transformed feats
    float*    x1    = hbuf  + (size_t)N * 128;            // [N,128] layer-1 output
    float*    asrc  = x1    + (size_t)N * 128;            // [N,8]
    float*    adst  = asrc  + (size_t)N * 8;              // [N,8]
    unsigned* emax  = (unsigned*)(adst + (size_t)N * 8);  // [N,8] encoded seg-max
    float*    denom = (float*)(emax + (size_t)N * 8);     // [N,8]
    float*    ebuf  = denom + (size_t)N * 8;              // [E,8] logits -> exp
    _Float16* Afrag = (_Float16*)(ebuf + (size_t)E * 8);  // [N*128] f16 A-fragments
    _Float16* Bfrag = Afrag + (size_t)((N + 15) / 16) * 16 * 128; // [128*128] f16

    // Output layout: x2 [N*128] | alpha1 [E*8] | alpha2 [E*8]
    float* x2_out     = (float*)d_out;
    float* alpha1_out = x2_out + (size_t)N * 128;
    float* alpha2_out = alpha1_out + (size_t)E * 8;

    // Layer 1: x -> x1 (workspace), alpha1 -> d_out
    run_gat_layer(x, W1, a_src1, a_dst1, b1, src, dst, Afrag, Bfrag,
                  hbuf, asrc, adst, emax, denom, ebuf,
                  x1, alpha1_out, N, E, stream);

    // Layer 2: x1 -> x2 (d_out), alpha2 -> d_out
    run_gat_layer(x1, W2, a_src2, a_dst2, b2, src, dst, Afrag, Bfrag,
                  hbuf, asrc, adst, emax, denom, ebuf,
                  x2_out, alpha2_out, N, E, stream);
}
